// GAT_18854906429892
// MI455X (gfx1250) — matrix operand order repaired and checked by reading, not compiled, not run on hardware
//
#include <hip/hip_runtime.h>
#include <hip/hip_bf16.h>
#include <cstdint>

// ---------------------------------------------------------------------------
// GAT (2-layer) for MI455X / gfx1250, wave32.
// GEMMs: TDM (tensor_load_to_lds) stages the 16xK A tile into LDS (with DMA
//        row padding to kill bank conflicts), 1 wave per 16x16 C tile consumes
//        it with v_wmma_f32_16x16x32_bf16; B panels stream from global/L2.
// Edge phase: 3-pass segment softmax + wave-per-edge weighted scatter-add.
// ---------------------------------------------------------------------------

typedef __attribute__((ext_vector_type(16))) __bf16 v16bf;
typedef __attribute__((ext_vector_type(8)))  float  v8f;
typedef __attribute__((ext_vector_type(4)))  unsigned int v4u;
typedef __attribute__((ext_vector_type(8)))  int   v8i;
typedef __attribute__((ext_vector_type(4)))  int   v4i;

constexpr int   Nn       = 100000;
constexpr int   Ne       = 1600000;
constexpr int   EP       = Ne + Nn;      // edges + self loops
constexpr int   IN_DIM   = 256;
constexpr int   HIDDEN   = 128;
constexpr int   OUT_DIM  = 64;
constexpr int   HEADS    = 2;
constexpr float NEG_SLOPE = 0.2f;

#define DEVI __device__ __forceinline__

DEVI unsigned short f2bf(float f) {
  unsigned u = __float_as_uint(f);
  unsigned r = 0x7FFFu + ((u >> 16) & 1u);           // round-to-nearest-even
  return (unsigned short)((u + r) >> 16);
}
// Order-preserving float<->uint mapping for atomicMax-based segment max.
DEVI unsigned f2mono(float f) {
  unsigned u = __float_as_uint(f);
  return (u & 0x80000000u) ? ~u : (u | 0x80000000u);
}
DEVI float mono2f(unsigned u) {
  return __uint_as_float((u & 0x80000000u) ? (u & 0x7FFFFFFFu) : ~u);
}

// --------------------------------- utility ---------------------------------

__global__ void k_zero_u32(unsigned* p, long long n) {
  long long i = (long long)blockIdx.x * blockDim.x + threadIdx.x;
  if (i < n) p[i] = 0u;
}

__global__ void k_cvt_bf16(const float* __restrict__ in,
                           unsigned short* __restrict__ out, long long n) {
  long long i = (long long)blockIdx.x * blockDim.x + threadIdx.x;
  if (i < n) out[i] = f2bf(in[i]);
}

// W[K][Ncols] row-major f32  ->  Wt[Ncols][K] bf16 (K-contiguous per column)
__global__ void k_cvt_transpose(const float* __restrict__ w,
                                unsigned short* __restrict__ wt,
                                int K, int Ncols) {
  long long i = (long long)blockIdx.x * blockDim.x + threadIdx.x;
  if (i >= (long long)K * Ncols) return;
  int k = (int)(i / Ncols), n = (int)(i % Ncols);
  wt[(size_t)n * K + k] = f2bf(w[i]);
}

// ------------------------------- WMMA GEMM ---------------------------------
// One workgroup = one 16-row output strip (tm). Wave 0 issues a single TDM
// descriptor moving the 16xK bf16 A tile into LDS with a 16B pad after each
// row (LDS row stride = K+8 elems -> conflict-free ds_load_b128). Each wave w
// computes D[tm*16 .. +15][w*16 .. +15] with v_wmma_f32_16x16x32_bf16,
// streaming its K-contiguous B column panel from global (weights sit in L2).
//
// VGPR layouts per CDNA5 ISA 7.12.2:
//   A 16x32 : lane m=l&15; elems 0..7 -> K = half*8+e ; elems 8..15 -> K=16+half*8+(e-8)
//   B 32x16 : lane n=l&15; elems 0..7 -> K = half*8+e ; elems 8..15 -> K=16+half*8+(e-8) (same as A)
//   D 16x16 : lane n=l&15; vgpr i -> M = half*8+i
__global__ void k_wmma_gemm_tdm(const unsigned short* __restrict__ A,
                                const unsigned short* __restrict__ Bt,
                                float* __restrict__ D, int K, int Nout) {
  extern __shared__ unsigned short sA[];   // [16][K+8]
  const int tm   = blockIdx.x;
  const int wave = threadIdx.x >> 5;       // = tn tile
  const int lane = threadIdx.x & 31;
  const int half = lane >> 4;
  const int l15  = lane & 15;
  const int RS   = K + 8;                  // padded LDS row stride (elements)

  if (wave == 0) {
    // ---- Tensor DMA descriptor (D#), ISA 8.3/8.4 ----
    unsigned long long ga = (unsigned long long)(uintptr_t)(A + (size_t)tm * 16 * K);
    unsigned ldsOff = (unsigned)(unsigned long long)(uintptr_t)sA;  // low 32 = LDS offset
    unsigned lgK    = 31u - (unsigned)__clz(K);
    // group0: count=1 | lds_addr | global_addr[56:0] | type=2
    v4u g0 = { 1u,
               ldsOff,
               (unsigned)(ga & 0xFFFFFFFFu),
               (unsigned)((ga >> 32) & 0x01FFFFFFu) | (2u << 30) };
    // group1: data_size=2B, pad_enable, pad_interval=K/2 DWORDs (one row),
    // pad_amount=4 DWORDs (16B); tensor = [Nn rows][K], tile = [16][K].
    v8i g1 = { (int)((1u << 16) | (1u << 20) | ((lgK - 2u) << 22) | (3u << 25)),
               (int)(((unsigned)K & 0xFFFFu) << 16),                      // tensor_dim0 lo16
               (int)((((unsigned)K >> 16) & 0xFFFFu) |
                     (((unsigned)Nn & 0xFFFFu) << 16)),                   // dim0 hi | dim1 lo
               (int)((((unsigned)Nn >> 16) & 0xFFFFu) |
                     (((unsigned)K & 0xFFFFu) << 16)),                    // dim1 hi | tile_dim0
               16,                                                        // tile_dim1=16, tile_dim2=0
               K,                                                         // tensor_dim0_stride lo32
               0, 0 };
    v4i g2 = { 0, 0, 0, 0 };
    v4i g3 = { 0, 0, 0, 0 };
    v8i g4 = { 0, 0, 0, 0, 0, 0, 0, 0 };   // extra group (clang-23 6-arg form)
    __builtin_amdgcn_tensor_load_to_lds(g0, g1, g2, g3, g4, /*cpol=*/0);
    __builtin_amdgcn_s_wait_tensorcnt(0);
  }
  __syncthreads();

  const unsigned short* arow = sA + l15 * RS;
  const unsigned short* bcol = Bt + (size_t)(wave * 16 + l15) * K;

  union U16 { uint4 q[2]; v16bf v; };
  v8f acc = {};
  for (int k = 0; k < K; k += 32) {
    U16 a, b;
    a.q[0] = *(const uint4*)(arow + k + half * 8);        // LDS (ds_load_b128)
    a.q[1] = *(const uint4*)(arow + k + 16 + half * 8);
    b.q[0] = *(const uint4*)(bcol + k + half * 8);        // global (L2-resident)
    b.q[1] = *(const uint4*)(bcol + k + 16 + half * 8);
    acc = __builtin_amdgcn_wmma_f32_16x16x32_bf16(
        /*neg_a=*/false, a.v, /*neg_b=*/false, b.v,
        /*c_mod=*/(short)0, acc, /*reuse_a=*/false, /*reuse_b=*/false);
  }

  union UC { v8f v; float f[8]; } c;
  c.v = acc;
  float* drow = D + (size_t)(tm * 16 + half * 8) * Nout + wave * 16 + l15;
#pragma unroll
  for (int i = 0; i < 8; ++i) drow[(size_t)i * Nout] = c.f[i];
}

// --------------------------- per-node attention logits ----------------------
// One wave per node: a_s[n,h] = dot(feat[n,h,:], att_s[h,:]); same for a_d.
__global__ void k_logits(const float* __restrict__ feat,
                         const float* __restrict__ att_s,
                         const float* __restrict__ att_d,
                         float* __restrict__ as_, float* __restrict__ ad_,
                         int H, int C, int nNodes) {
  int wid  = (int)(((long long)blockIdx.x * blockDim.x + threadIdx.x) >> 5);
  int lane = threadIdx.x & 31;
  if (wid >= nNodes) return;
  const float* f = feat + (size_t)wid * H * C;
  for (int h = 0; h < H; ++h) {
    float s = 0.f, d = 0.f;
    for (int c = lane; c < C; c += 32) {
      float v = f[h * C + c];
      s += v * att_s[h * C + c];
      d += v * att_d[h * C + c];
    }
    for (int off = 16; off; off >>= 1) {
      s += __shfl_xor(s, off, 32);
      d += __shfl_xor(d, off, 32);
    }
    if (lane == 0) {
      as_[(size_t)wid * H + h] = s;
      ad_[(size_t)wid * H + h] = d;
    }
  }
}

// ------------------------------- edge passes --------------------------------

DEVI void edge_ends(const long long* __restrict__ ei, int i, long long& s, long long& d) {
  if (i < Ne) { s = ei[i]; d = ei[(size_t)Ne + i]; }
  else        { s = d = i - Ne; }                    // self loop
}

// pass 1: e = leaky_relu(a_s[src]+a_d[dst]); segment max into mbuf (mono u32)
__global__ void k_edge_scores(const long long* __restrict__ ei,
                              const float* __restrict__ as_,
                              const float* __restrict__ ad_,
                              float* __restrict__ ebuf,
                              unsigned* __restrict__ mbuf, int H) {
  int i = (int)((long long)blockIdx.x * blockDim.x + threadIdx.x);
  if (i >= EP) return;
  long long s, d;
  edge_ends(ei, i, s, d);
  for (int h = 0; h < H; ++h) {
    float e = as_[s * H + h] + ad_[d * H + h];
    e = e > 0.f ? e : NEG_SLOPE * e;
    ebuf[(size_t)i * H + h] = e;
    atomicMax(&mbuf[d * H + h], f2mono(e));
  }
}

// pass 2: alpha = exp(e - m[dst]) (stored over ebuf); denom[dst] += alpha
__global__ void k_edge_exp(const long long* __restrict__ ei,
                           float* __restrict__ ebuf,
                           const unsigned* __restrict__ mbuf,
                           float* __restrict__ den, int H) {
  int i = (int)((long long)blockIdx.x * blockDim.x + threadIdx.x);
  if (i >= EP) return;
  long long s, d;
  edge_ends(ei, i, s, d);
  for (int h = 0; h < H; ++h) {
    float a = __expf(ebuf[(size_t)i * H + h] - mono2f(mbuf[d * H + h]));
    ebuf[(size_t)i * H + h] = a;
    atomicAdd(&den[d * H + h], a);
  }
}

// pass 3: acc[dst, h, :] += (alpha/denom) * feat[src, h, :]   (wave per edge)
// lgC = log2(C); channels-per-lane = (H*C)/32.
__global__ void k_aggregate(const long long* __restrict__ ei,
                            const float* __restrict__ alpha,
                            const float* __restrict__ den,
                            const float* __restrict__ feat,
                            float* __restrict__ acc, int H, int lgC) {
  int wid  = (int)(((long long)blockIdx.x * blockDim.x + threadIdx.x) >> 5);
  int lane = threadIdx.x & 31;
  if (wid >= EP) return;
  long long s, d;
  edge_ends(ei, wid, s, d);
  const int CH  = H << lgC;
  const int per = CH >> 5;                 // 8 (layer1) or 2 (layer2) chans/lane
  const float* fs = feat + (size_t)s * CH;
  float*       ad = acc  + (size_t)d * CH;
  __builtin_prefetch(fs + lane * per, 0, 3);   // global_prefetch_b8
  for (int j = 0; j < per; ++j) {
    int c = lane * per + j;
    int h = c >> lgC;
    float w = alpha[(size_t)wid * H + h] / den[d * H + h];
    atomicAdd(&ad[c], w * fs[c]);
  }
}

// layer-1 finalize: mean over heads + bias + ReLU, emit bf16 for GEMM2
__global__ void k_final1(const float* __restrict__ acc,
                         const float* __restrict__ b1,
                         unsigned short* __restrict__ hb) {
  long long i = (long long)blockIdx.x * blockDim.x + threadIdx.x;
  if (i >= (long long)Nn * HIDDEN) return;
  int n = (int)(i >> 7), c = (int)(i & 127);
  float v = (acc[(size_t)n * 256 + c] + acc[(size_t)n * 256 + 128 + c]) * 0.5f + b1[c];
  hb[i] = f2bf(fmaxf(v, 0.f));
}

// layer-2 finalize: add bias in place on d_out
__global__ void k_bias2(float* __restrict__ out, const float* __restrict__ b2) {
  long long i = (long long)blockIdx.x * blockDim.x + threadIdx.x;
  if (i >= (long long)Nn * OUT_DIM) return;
  out[i] += b2[i & 63];
}

// --------------------------------- driver -----------------------------------

static inline int cdiv(long long a, long long b) { return (int)((a + b - 1) / b); }

extern "C" void kernel_launch(void* const* d_in, const int* in_sizes, int n_in,
                              void* d_out, int out_size, void* d_ws, size_t ws_size,
                              hipStream_t stream) {
  (void)in_sizes; (void)n_in; (void)out_size; (void)ws_size;
  const float*     x    = (const float*)d_in[0];
  const long long* ei   = (const long long*)d_in[1];   // int64 [2][E]
  const float*     W1   = (const float*)d_in[2];
  const float*     asw1 = (const float*)d_in[3];
  const float*     adw1 = (const float*)d_in[4];
  const float*     b1   = (const float*)d_in[5];
  const float*     W2   = (const float*)d_in[6];
  const float*     asw2 = (const float*)d_in[7];
  const float*     adw2 = (const float*)d_in[8];
  const float*     b2   = (const float*)d_in[9];
  float*           out  = (float*)d_out;

  // workspace carve-up (256B aligned)
  char* p = (char*)d_ws;
  auto take = [&](size_t bytes) -> char* {
    char* r = p; p += (bytes + 255) & ~(size_t)255; return r;
  };
  unsigned short* xb   = (unsigned short*)take((size_t)Nn * IN_DIM * 2);
  unsigned short* w1t  = (unsigned short*)take((size_t)IN_DIM * HEADS * HIDDEN * 2);
  float*          h1   = (float*)take((size_t)Nn * HEADS * HIDDEN * 4);
  float*          as1  = (float*)take((size_t)Nn * HEADS * 4);
  float*          ad1  = (float*)take((size_t)Nn * HEADS * 4);
  unsigned*       m1   = (unsigned*)take((size_t)Nn * HEADS * 4);
  float*          den1 = (float*)take((size_t)Nn * HEADS * 4);
  float*          e1   = (float*)take((size_t)EP * HEADS * 4);
  float*          acc1 = (float*)take((size_t)Nn * HEADS * HIDDEN * 4);
  unsigned short* hb   = (unsigned short*)take((size_t)Nn * HIDDEN * 2);
  unsigned short* w2t  = (unsigned short*)take((size_t)HIDDEN * OUT_DIM * 2);
  float*          h2   = (float*)take((size_t)Nn * OUT_DIM * 4);
  float*          as2  = (float*)take((size_t)Nn * 4);
  float*          ad2  = (float*)take((size_t)Nn * 4);
  unsigned*       m2   = (unsigned*)take((size_t)Nn * 4);
  float*          den2 = (float*)take((size_t)Nn * 4);
  float*          e2   = (float*)take((size_t)EP * 4);

  const int TB = 256;

  // ---- init accumulators / maxima (mono(0)=below all reals) / output ----
  k_zero_u32<<<cdiv((long long)Nn * HEADS, TB), TB, 0, stream>>>(m1, (long long)Nn * HEADS);
  k_zero_u32<<<cdiv((long long)Nn * HEADS, TB), TB, 0, stream>>>((unsigned*)den1, (long long)Nn * HEADS);
  k_zero_u32<<<cdiv((long long)Nn * HEADS * HIDDEN, TB), TB, 0, stream>>>((unsigned*)acc1, (long long)Nn * HEADS * HIDDEN);
  k_zero_u32<<<cdiv((long long)Nn, TB), TB, 0, stream>>>(m2, (long long)Nn);
  k_zero_u32<<<cdiv((long long)Nn, TB), TB, 0, stream>>>((unsigned*)den2, (long long)Nn);
  k_zero_u32<<<cdiv((long long)Nn * OUT_DIM, TB), TB, 0, stream>>>((unsigned*)out, (long long)Nn * OUT_DIM);

  // ---- layer 1: GEMM (TDM -> LDS -> bf16 WMMA) ----
  k_cvt_bf16<<<cdiv((long long)Nn * IN_DIM, TB), TB, 0, stream>>>(x, xb, (long long)Nn * IN_DIM);
  k_cvt_transpose<<<cdiv((long long)IN_DIM * HEADS * HIDDEN, TB), TB, 0, stream>>>(W1, w1t, IN_DIM, HEADS * HIDDEN);
  k_wmma_gemm_tdm<<<Nn / 16, ((HEADS * HIDDEN) / 16) * 32,
                    (size_t)16 * (IN_DIM + 8) * 2, stream>>>(xb, w1t, h1, IN_DIM, HEADS * HIDDEN);

  // ---- layer 1: attention + segment softmax + aggregation ----
  k_logits<<<cdiv((long long)Nn * 32, TB), TB, 0, stream>>>(h1, asw1, adw1, as1, ad1, HEADS, HIDDEN, Nn);
  k_edge_scores<<<cdiv(EP, TB), TB, 0, stream>>>(ei, as1, ad1, e1, m1, HEADS);
  k_edge_exp<<<cdiv(EP, TB), TB, 0, stream>>>(ei, e1, m1, den1, HEADS);
  k_aggregate<<<cdiv((long long)EP * 32, TB), TB, 0, stream>>>(ei, e1, den1, h1, acc1, HEADS, 7);
  k_final1<<<cdiv((long long)Nn * HIDDEN, TB), TB, 0, stream>>>(acc1, b1, hb);

  // ---- layer 2: GEMM (TDM -> LDS -> bf16 WMMA) ----
  k_cvt_transpose<<<cdiv((long long)HIDDEN * OUT_DIM, TB), TB, 0, stream>>>(W2, w2t, HIDDEN, OUT_DIM);
  k_wmma_gemm_tdm<<<Nn / 16, (OUT_DIM / 16) * 32,
                    (size_t)16 * (HIDDEN + 8) * 2, stream>>>(hb, w2t, h2, HIDDEN, OUT_DIM);

  // ---- layer 2: attention + segment softmax + aggregation (into d_out) ----
  k_logits<<<cdiv((long long)Nn * 32, TB), TB, 0, stream>>>(h2, asw2, adw2, as2, ad2, 1, OUT_DIM, Nn);
  k_edge_scores<<<cdiv(EP, TB), TB, 0, stream>>>(ei, as2, ad2, e2, m2, 1);
  k_edge_exp<<<cdiv(EP, TB), TB, 0, stream>>>(ei, e2, m2, den2, 1);
  k_aggregate<<<cdiv((long long)EP * 32, TB), TB, 0, stream>>>(ei, e2, den2, h2, out, 1, 6);
  k_bias2<<<cdiv((long long)Nn * OUT_DIM, TB), TB, 0, stream>>>(out, b2);
}
